// MLPExperts_5265629904928
// MI455X (gfx1250) — matrix operand order repaired and checked by reading, not compiled, not run on hardware
//
#include <hip/hip_runtime.h>
#include <math.h>

// MoE MLP: out = (GELU(x @ c_fc + fc_bias)) @ c_proj + proj_bias, per expert.
// Two bf16-WMMA GEMM kernels, double-buffered LDS software pipeline,
// intermediate h kept as bf16 in d_ws (128 MB).
// fp32->bf16 conversion via V_FMA_MIXLO/MIXHI_BF16 (1 VALU op per element,
// RNE, packs two elements into one VGPR).

#define N_EXP  8
#define N_EMBD 1024
#define D_FF   4096
#define TOK    2048

#define BM 128
#define BN 128
#define BK 32
#define LDSS 48   // padded LDS row stride (elements); 96B keeps 16B align, rotates banks

typedef __attribute__((ext_vector_type(16))) __bf16 v16bf;
typedef __attribute__((ext_vector_type(8)))  float  v8f;

union FragU { v16bf v; uint4 q[2]; };

// fp32 -> bf16 bits, RNE, manual fallback (no hw cvt available)
__device__ __forceinline__ unsigned short f2bf_sw(float x) {
  unsigned int u = __float_as_uint(x);
  return (unsigned short)((u + 0x7FFFu + ((u >> 16) & 1u)) >> 16);
}

// pack two fp32 -> bf16x2 in one u32
__device__ __forceinline__ unsigned int pack2_bf16(float a, float b) {
#if __has_builtin(__builtin_amdgcn_cvt_pk_bf16_f32)
  auto p = __builtin_amdgcn_cvt_pk_bf16_f32(a, b);
  unsigned int r;
  __builtin_memcpy(&r, &p, 4);
  return r;
#else
  // V_FMA_MIXLO_BF16 / V_FMA_MIXHI_BF16 (CDNA5 VOP3P op 62/63):
  // D[15:0]/D[31:16] = bf16(fma(f32 src0, 1.0, 0)), other half preserved.
  unsigned int r;
  asm("v_fma_mixlo_bf16 %0, %1, 1.0, 0\n\t"
      "v_fma_mixhi_bf16 %0, %2, 1.0, 0"
      : "=&v"(r)
      : "v"(a), "v"(b));
  return r;
#endif
}

// single fp32 -> bf16 bits via hardware mix-convert
__device__ __forceinline__ unsigned short f2bf(float a) {
#if __has_builtin(__builtin_amdgcn_cvt_pk_bf16_f32)
  return (unsigned short)pack2_bf16(a, 0.0f);
#else
  unsigned int r;
  asm("v_fma_mixlo_bf16 %0, %1, 1.0, 0" : "=v"(r) : "v"(a));
  return (unsigned short)r;
#endif
}

__device__ __forceinline__ float gelu_exact(float v) {
  return 0.5f * v * (1.0f + erff(v * 0.70710678118654752f));
}

// ---------------------------------------------------------------------------
// GEMM1: h[e] = GELU(x[e] @ c_fc[e] + fc_bias[e]) -> bf16 workspace.
// grid: (D_FF/BN, TOK/BM, N_EXP), block 256 (8 wave32)
// ---------------------------------------------------------------------------
__global__ __launch_bounds__(256)
void mlp_gemm1_gelu(const float* __restrict__ X, const float* __restrict__ W,
                    const float* __restrict__ BIAS, unsigned short* __restrict__ H) {
  __shared__ unsigned short lA[2][BM * LDSS];   // [m][k] bf16
  __shared__ unsigned short lB[2][BN * LDSS];   // [n][k] bf16 (transposed)

  const int e    = blockIdx.z;
  const int tm   = blockIdx.y;
  const int tn   = blockIdx.x;
  const int tid  = threadIdx.x;
  const int lane = tid & 31;
  const int wave = tid >> 5;
  const int half = lane >> 4;
  const int ln16 = lane & 15;
  const int wm   = wave & 3;      // rows wm*32 (2 m-tiles)
  const int wn   = wave >> 2;     // cols wn*64 (4 n-tiles)

  const float* x  = X + (size_t)e * TOK * N_EMBD + (size_t)tm * BM * N_EMBD;
  const float* w  = W + (size_t)e * N_EMBD * D_FF + (size_t)tn * BN;
  const float* bb = BIAS + (size_t)e * D_FF + (size_t)tn * BN;
  unsigned short* h = H + (size_t)e * TOK * D_FF + (size_t)tm * BM * D_FF
                        + (size_t)tn * BN;

  v8f acc[2][4] = {};
  float4 ra[4], rb[4];

  auto globalLoad = [&](int k0) {
#pragma unroll
    for (int i = 0; i < 4; ++i) {
      int f = tid + i * 256;
      ra[i] = *(const float4*)(x + (size_t)(f >> 3) * N_EMBD + k0 + ((f & 7) << 2));
      rb[i] = *(const float4*)(w + (size_t)(k0 + (f >> 5)) * D_FF + ((f & 31) << 2));
    }
  };
  auto ldsStore = [&](int buf) {
#pragma unroll
    for (int i = 0; i < 4; ++i) {
      int f = tid + i * 256;
      {
        int row = f >> 3, c = (f & 7) << 2;
        uint2 p;
        p.x = pack2_bf16(ra[i].x, ra[i].y);
        p.y = pack2_bf16(ra[i].z, ra[i].w);
        *(uint2*)(&lA[buf][row * LDSS + c]) = p;          // ds_store_b64
      }
      {
        int kr = f >> 5, c = (f & 31) << 2;
        unsigned int p01 = pack2_bf16(rb[i].x, rb[i].y);
        unsigned int p23 = pack2_bf16(rb[i].z, rb[i].w);
        lB[buf][(c + 0) * LDSS + kr] = (unsigned short)p01;
        lB[buf][(c + 1) * LDSS + kr] = (unsigned short)(p01 >> 16);
        lB[buf][(c + 2) * LDSS + kr] = (unsigned short)p23;
        lB[buf][(c + 3) * LDSS + kr] = (unsigned short)(p23 >> 16);
      }
    }
  };
  auto computeStep = [&](int buf) {
    FragU a[2], b[4];
#pragma unroll
    for (int mt = 0; mt < 2; ++mt) {
      int m = wm * 32 + mt * 16 + ln16;
      // A 16-bit layout: V0-3 -> K = half*8 + 0..7 ; V4-7 -> K = 16 + half*8 + 0..7
      a[mt].q[0] = *(const uint4*)(&lA[buf][m * LDSS + half * 8]);
      a[mt].q[1] = *(const uint4*)(&lA[buf][m * LDSS + 16 + half * 8]);
    }
#pragma unroll
    for (int nt = 0; nt < 4; ++nt) {
      int n = wn * 64 + nt * 16 + ln16;
      // B layout (col n per lane): V0-3 -> K = half*8 + 0..7 ; V4-7 -> K = 16 + half*8 + 0..7
      b[nt].q[0] = *(const uint4*)(&lB[buf][n * LDSS + half * 8]);
      b[nt].q[1] = *(const uint4*)(&lB[buf][n * LDSS + 16 + half * 8]);
    }
#pragma unroll
    for (int mt = 0; mt < 2; ++mt)
#pragma unroll
      for (int nt = 0; nt < 4; ++nt)
        acc[mt][nt] = __builtin_amdgcn_wmma_f32_16x16x32_bf16(
            false, a[mt].v, false, b[nt].v, (short)0, acc[mt][nt], false, false);
  };

  const int NK = N_EMBD / BK;
  globalLoad(0);
  ldsStore(0);
  __syncthreads();
#pragma unroll 2
  for (int k = 0; k < NK; ++k) {
    int cur = k & 1;
    if (k + 1 < NK) globalLoad((k + 1) * BK);   // HBM loads overlap WMMA below
    computeStep(cur);
    if (k + 1 < NK) ldsStore(cur ^ 1);
    __syncthreads();
  }

  // epilogue: bias + exact GELU -> bf16
#pragma unroll
  for (int nt = 0; nt < 4; ++nt) {
    int n = wn * 64 + nt * 16 + ln16;
    float bias = bb[n];
#pragma unroll
    for (int mt = 0; mt < 2; ++mt) {
#pragma unroll
      for (int r = 0; r < 8; ++r) {
        int m = wm * 32 + mt * 16 + half * 8 + r;  // C/D layout: M = r + 8*half
        float v = acc[mt][nt][r] + bias;
        h[(size_t)m * D_FF + n] = f2bf(gelu_exact(v));
      }
    }
  }
}

// ---------------------------------------------------------------------------
// GEMM2: out[e] = h[e] @ c_proj[e] + proj_bias[e], fp32 output.
// grid: (N_EMBD/BN, TOK/BM, N_EXP), block 256
// ---------------------------------------------------------------------------
__global__ __launch_bounds__(256)
void mlp_gemm2(const unsigned short* __restrict__ H, const float* __restrict__ W,
               const float* __restrict__ BIAS, float* __restrict__ OUT) {
  __shared__ unsigned short lA[2][BM * LDSS];
  __shared__ unsigned short lB[2][BN * LDSS];

  const int e    = blockIdx.z;
  const int tm   = blockIdx.y;
  const int tn   = blockIdx.x;
  const int tid  = threadIdx.x;
  const int lane = tid & 31;
  const int wave = tid >> 5;
  const int half = lane >> 4;
  const int ln16 = lane & 15;
  const int wm   = wave & 3;
  const int wn   = wave >> 2;

  const unsigned short* hsrc = H + (size_t)e * TOK * D_FF + (size_t)tm * BM * D_FF;
  const float* w  = W + (size_t)e * D_FF * N_EMBD + (size_t)tn * BN;
  const float* bb = BIAS + (size_t)e * N_EMBD + (size_t)tn * BN;
  float* out = OUT + (size_t)e * TOK * N_EMBD + (size_t)tm * BM * N_EMBD
                   + (size_t)tn * BN;

  v8f acc[2][4] = {};
  uint4  ra[2];     // A already bf16: raw 8-element groups
  float4 rb[4];

  auto globalLoad = [&](int k0) {
#pragma unroll
    for (int i = 0; i < 2; ++i) {
      int f = tid + i * 256;                       // 0..511
      ra[i] = *(const uint4*)(hsrc + (size_t)(f >> 2) * D_FF + k0 + ((f & 3) << 3));
    }
#pragma unroll
    for (int i = 0; i < 4; ++i) {
      int f = tid + i * 256;
      rb[i] = *(const float4*)(w + (size_t)(k0 + (f >> 5)) * N_EMBD + ((f & 31) << 2));
    }
  };
  auto ldsStore = [&](int buf) {
#pragma unroll
    for (int i = 0; i < 2; ++i) {
      int f = tid + i * 256;
      *(uint4*)(&lA[buf][(f >> 2) * LDSS + ((f & 3) << 3)]) = ra[i];  // ds_store_b128
    }
#pragma unroll
    for (int i = 0; i < 4; ++i) {
      int f = tid + i * 256;
      int kr = f >> 5, c = (f & 31) << 2;
      unsigned int p01 = pack2_bf16(rb[i].x, rb[i].y);
      unsigned int p23 = pack2_bf16(rb[i].z, rb[i].w);
      lB[buf][(c + 0) * LDSS + kr] = (unsigned short)p01;
      lB[buf][(c + 1) * LDSS + kr] = (unsigned short)(p01 >> 16);
      lB[buf][(c + 2) * LDSS + kr] = (unsigned short)p23;
      lB[buf][(c + 3) * LDSS + kr] = (unsigned short)(p23 >> 16);
    }
  };
  auto computeStep = [&](int buf) {
    FragU a[2], b[4];
#pragma unroll
    for (int mt = 0; mt < 2; ++mt) {
      int m = wm * 32 + mt * 16 + ln16;
      a[mt].q[0] = *(const uint4*)(&lA[buf][m * LDSS + half * 8]);
      a[mt].q[1] = *(const uint4*)(&lA[buf][m * LDSS + 16 + half * 8]);
    }
#pragma unroll
    for (int nt = 0; nt < 4; ++nt) {
      int n = wn * 64 + nt * 16 + ln16;
      b[nt].q[0] = *(const uint4*)(&lB[buf][n * LDSS + half * 8]);
      b[nt].q[1] = *(const uint4*)(&lB[buf][n * LDSS + 16 + half * 8]);
    }
#pragma unroll
    for (int mt = 0; mt < 2; ++mt)
#pragma unroll
      for (int nt = 0; nt < 4; ++nt)
        acc[mt][nt] = __builtin_amdgcn_wmma_f32_16x16x32_bf16(
            false, a[mt].v, false, b[nt].v, (short)0, acc[mt][nt], false, false);
  };

  const int NK = D_FF / BK;
  globalLoad(0);
  ldsStore(0);
  __syncthreads();
#pragma unroll 2
  for (int k = 0; k < NK; ++k) {
    int cur = k & 1;
    if (k + 1 < NK) globalLoad((k + 1) * BK);
    computeStep(cur);
    if (k + 1 < NK) ldsStore(cur ^ 1);
    __syncthreads();
  }

#pragma unroll
  for (int nt = 0; nt < 4; ++nt) {
    int n = wn * 64 + nt * 16 + ln16;
    float bias = bb[n];
#pragma unroll
    for (int mt = 0; mt < 2; ++mt) {
#pragma unroll
      for (int r = 0; r < 8; ++r) {
        int m = wm * 32 + mt * 16 + half * 8 + r;
        out[(size_t)m * N_EMBD + n] = acc[mt][nt][r] + bias;
      }
    }
  }
}

extern "C" void kernel_launch(void* const* d_in, const int* in_sizes, int n_in,
                              void* d_out, int out_size, void* d_ws, size_t ws_size,
                              hipStream_t stream) {
  const float* x         = (const float*)d_in[0];  // [8, 2048, 1024]
  const float* c_fc      = (const float*)d_in[1];  // [8, 1024, 4096]
  const float* c_proj    = (const float*)d_in[2];  // [8, 4096, 1024]
  const float* fc_bias   = (const float*)d_in[3];  // [8, 1, 4096]
  const float* proj_bias = (const float*)d_in[4];  // [8, 1, 1024]
  float* out = (float*)d_out;                      // [8, 2048, 1024] fp32
  unsigned short* h = (unsigned short*)d_ws;       // [8, 2048, 4096] bf16 = 128 MB

  dim3 g1(D_FF / BN, TOK / BM, N_EXP);    // 32 x 16 x 8
  mlp_gemm1_gelu<<<g1, 256, 0, stream>>>(x, c_fc, fc_bias, h);

  dim3 g2(N_EMBD / BN, TOK / BM, N_EXP);  // 8 x 16 x 8
  mlp_gemm2<<<g2, 256, 0, stream>>>(h, c_proj, proj_bias, out);
}
